// MSA_Conv_90366111908149
// MI455X (gfx1250) — matrix operand order repaired and checked by reading, not compiled, not run on hardware
//
#include <hip/hip_runtime.h>
#include <hip/hip_fp16.h>

typedef __attribute__((ext_vector_type(16))) _Float16 v16h;
typedef __attribute__((ext_vector_type(8)))  _Float16 v8h;
typedef __attribute__((ext_vector_type(8)))  float    v8f;

#define T_  4
#define N_  512
#define C_  256
#define V_  25
#define VP  32
#define H_  8
#define D_  32
#define XS_STRIDE 264   // halfs; 528B row stride (16B aligned), bank-spread
#define AT_STRIDE 40    // halfs; 80B row stride (16B aligned)
#define OS_STRIDE 26    // floats; output staging stride

union Frag16 { v16h v; v8h h[2]; };

__device__ __forceinline__ v8f wmma16(const Frag16& a, const Frag16& b, v8f c) {
  return __builtin_amdgcn_wmma_f32_16x16x32_f16(false, a.v, false, b.v,
                                                (short)0, c, false, false);
}

// LIF: v = (v + u)/2 ; spike = (v >= vth) ; hard reset to 0
__device__ __forceinline__ float lif_step(float& vm, float u, float vth) {
  float vn = 0.5f * (vm + u);
  float s  = (vn >= vth) ? 1.0f : 0.0f;
  vm       = (vn >= vth) ? 0.0f : vn;
  return s;
}

// ---------------------------------------------------------------------------
// K0a: fold eval-mode BN (and optional conv bias) into f16 weights + f32 bias
// ---------------------------------------------------------------------------
__global__ void fold_bn_kernel(const float* __restrict__ w,
                               const float* __restrict__ gamma,
                               const float* __restrict__ beta,
                               const float* __restrict__ mean,
                               const float* __restrict__ var,
                               const float* __restrict__ convb, // nullptr if none
                               _Float16* __restrict__ w16,
                               float* __restrict__ bias) {
  int idx = blockIdx.x * blockDim.x + threadIdx.x;
  if (idx < C_ * C_) {
    int o = idx / C_;
    float inv = gamma[o] * rsqrtf(var[o] + 1e-5f);
    w16[idx] = (_Float16)(w[idx] * inv);
  }
  if (idx < C_) {
    float inv = gamma[idx] * rsqrtf(var[idx] + 1e-5f);
    float cb  = convb ? convb[idx] : 0.0f;
    bias[idx] = inv * (cb - mean[idx]) + beta[idx];
  }
}

// ---------------------------------------------------------------------------
// K0b: x [T,N,C,V] f32  ->  xT [T,N,VP,C] f16, zero-padded rows V_..VP-1.
// One block per (n,t); coalesced load, LDS transpose, coalesced store.
// ---------------------------------------------------------------------------
__global__ __launch_bounds__(256) void transpose_x_kernel(
    const float* __restrict__ x, _Float16* __restrict__ xT) {
  __shared__ __align__(16) _Float16 Xs[V_ * XS_STRIDE];
  const int n = blockIdx.x;
  const int t = blockIdx.y;
  const int tid = threadIdx.x;

  const float* __restrict__ xp = x + (((size_t)t * N_ + n) * C_) * V_;
  for (int idx = tid; idx < C_ * V_; idx += 256) {
    int c = idx / V_, v = idx - c * V_;
    Xs[v * XS_STRIDE + c] = (_Float16)xp[idx];
  }
  __syncthreads();
  _Float16* __restrict__ op = xT + (((size_t)t * N_ + n) * VP) * C_;
  for (int idx = tid; idx < VP * C_; idx += 256) {
    int v = idx / C_, c = idx - v * C_;
    op[idx] = (v < V_) ? Xs[v * XS_STRIDE + c] : (_Float16)0.0f;
  }
}

// ---------------------------------------------------------------------------
// K1: per (n, branch) fused conv1x1(+folded BN) + LIF over T. No LDS needed:
// A-frags (weights, t-invariant -> hoisted to VGPRs), B-frags straight from xT.
// Spikes -> s[t][n][v][c] f16, pad rows zeroed.
// ---------------------------------------------------------------------------
__global__ __launch_bounds__(256) void qkv_conv_lif_kernel(
    const _Float16* __restrict__ xT,
    const _Float16* __restrict__ wq16, const _Float16* __restrict__ wk16,
    const _Float16* __restrict__ wv16,
    const float* __restrict__ bq, const float* __restrict__ bk,
    const float* __restrict__ bv,
    _Float16* __restrict__ sq, _Float16* __restrict__ sk,
    _Float16* __restrict__ sv) {
  const int n  = blockIdx.x;
  const int br = blockIdx.y;
  const _Float16* __restrict__ W  = (br == 0) ? wq16 : ((br == 1) ? wk16 : wv16);
  const float*    __restrict__ Bb = (br == 0) ? bq   : ((br == 1) ? bk   : bv);
  _Float16*       __restrict__ S  = (br == 0) ? sq   : ((br == 1) ? sk   : sv);

  const int tid  = threadIdx.x;
  const int wave = tid >> 5;
  const int lane = tid & 31;
  const int g    = lane >> 4;
  const int ln   = lane & 15;
  const int m0[2] = { wave * 32, wave * 32 + 16 };

  float bias[2][8];
#pragma unroll
  for (int mt = 0; mt < 2; ++mt)
#pragma unroll
    for (int r = 0; r < 8; ++r) bias[mt][r] = Bb[m0[mt] + 8 * g + r];

  float vmem[2][2][8];
#pragma unroll
  for (int mt = 0; mt < 2; ++mt)
#pragma unroll
    for (int nt = 0; nt < 2; ++nt)
#pragma unroll
      for (int r = 0; r < 8; ++r) vmem[mt][nt][r] = 0.0f;

  for (int t = 0; t < T_; ++t) {
    const _Float16* __restrict__ xb = xT + (((size_t)t * N_ + n) * VP) * C_;
    if (t + 1 < T_) {  // one cacheline per thread covers the next 16KB tile
      const char* nx = (const char*)(xT + (((size_t)(t + 1) * N_ + n) * VP) * C_);
      __builtin_prefetch(nx + (size_t)tid * 64, 0, 1);
    }

    v8f acc[2][2] = {};
#pragma unroll
    for (int ks = 0; ks < C_ / 32; ++ks) {
      const int kb = ks * 32;
      Frag16 Af[2], Bf[2];
#pragma unroll
      for (int mt = 0; mt < 2; ++mt) {
        const _Float16* wr = W + (size_t)(m0[mt] + ln) * C_ + kb + 8 * g;
        Af[mt].h[0] = *(const v8h*)(wr);
        Af[mt].h[1] = *(const v8h*)(wr + 16);
      }
#pragma unroll
      for (int nt = 0; nt < 2; ++nt) {
        const _Float16* xr = xb + (size_t)(nt * 16 + ln) * C_ + kb + 8 * g;
        Bf[nt].h[0] = *(const v8h*)(xr);
        Bf[nt].h[1] = *(const v8h*)(xr + 16);
      }
#pragma unroll
      for (int mt = 0; mt < 2; ++mt)
#pragma unroll
        for (int nt = 0; nt < 2; ++nt)
          acc[mt][nt] = wmma16(Af[mt], Bf[nt], acc[mt][nt]);
    }

#pragma unroll
    for (int mt = 0; mt < 2; ++mt) {
#pragma unroll
      for (int nt = 0; nt < 2; ++nt) {
        const int v = nt * 16 + ln;
        v8h sp{};
#pragma unroll
        for (int r = 0; r < 8; ++r) {
          float u = acc[mt][nt][r] + bias[mt][r];
          float s = lif_step(vmem[mt][nt][r], u, 1.0f);
          sp[r] = (v < V_) ? (_Float16)s : (_Float16)0.0f;  // zero pad rows
        }
        _Float16* op = S + (((size_t)t * N_ + n) * VP + v) * C_ + m0[mt] + 8 * g;
        *(v8h*)op = sp;
      }
    }
  }
}

// ---------------------------------------------------------------------------
// K2: per (n,h), one wave: attn = q k^T, y = attn v * 0.125, LIF(0.5) over T.
// attnS LDS is reused to re-layout spikes for coalesced 64B/lane stores.
// ---------------------------------------------------------------------------
__global__ __launch_bounds__(32) void attn_lif_kernel(
    const _Float16* __restrict__ sq, const _Float16* __restrict__ sk,
    const _Float16* __restrict__ sv, _Float16* __restrict__ ys) {
  __shared__ __align__(16) _Float16 attnS[32 * AT_STRIDE];
  __shared__ __align__(16) _Float16 vT[32 * AT_STRIDE];

  const int n = blockIdx.x;
  const int h = blockIdx.y;
  const int lane = threadIdx.x;
  const int g = lane >> 4, ln = lane & 15;
  const int cb = h * D_;

  float vmem[2][2][8];
#pragma unroll
  for (int mt = 0; mt < 2; ++mt)
#pragma unroll
    for (int nt = 0; nt < 2; ++nt)
#pragma unroll
      for (int r = 0; r < 8; ++r) vmem[mt][nt][r] = 0.0f;

  for (int t = 0; t < T_; ++t) {
    const size_t base = (((size_t)t * N_ + n) * VP) * C_ + cb;

    // Q as A (rows v, K=d), K as B (cols w, K=d)
    Frag16 Aq[2], Bk[2];
#pragma unroll
    for (int mt = 0; mt < 2; ++mt) {
      const _Float16* p = sq + base + (size_t)(mt * 16 + ln) * C_ + 8 * g;
      Aq[mt].h[0] = *(const v8h*)p;
      Aq[mt].h[1] = *(const v8h*)(p + 16);
    }
#pragma unroll
    for (int nt = 0; nt < 2; ++nt) {
      const _Float16* p = sk + base + (size_t)(nt * 16 + ln) * C_ + 8 * g;
      Bk[nt].h[0] = *(const v8h*)p;
      Bk[nt].h[1] = *(const v8h*)(p + 16);
    }
    v8f aacc[2][2] = {};
#pragma unroll
    for (int mt = 0; mt < 2; ++mt)
#pragma unroll
      for (int nt = 0; nt < 2; ++nt)
        aacc[mt][nt] = wmma16(Aq[mt], Bk[nt], aacc[mt][nt]);

    __syncthreads();
    // attn (f32 accum layout) -> LDS f16 row-major [v][w]
#pragma unroll
    for (int mt = 0; mt < 2; ++mt)
#pragma unroll
      for (int nt = 0; nt < 2; ++nt)
#pragma unroll
        for (int r = 0; r < 8; ++r)
          attnS[(mt * 16 + 8 * g + r) * AT_STRIDE + nt * 16 + ln] =
              (_Float16)aacc[mt][nt][r];

    // V rows -> LDS transposed vT[d][w]
    {
      const _Float16* p = sv + base + (size_t)lane * C_;
#pragma unroll
      for (int dd = 0; dd < 4; ++dd) {
        v8h chunk = *(const v8h*)(p + dd * 8);
#pragma unroll
        for (int j = 0; j < 8; ++j)
          vT[(dd * 8 + j) * AT_STRIDE + lane] = chunk[j];
      }
    }
    __syncthreads();

    // y = attn @ v : A from attnS (rows v, K=w), B from vT (cols d, K=w)
    Frag16 Aa[2], Bv[2];
#pragma unroll
    for (int mt = 0; mt < 2; ++mt) {
      const _Float16* p = &attnS[(mt * 16 + ln) * AT_STRIDE + 8 * g];
      Aa[mt].h[0] = *(const v8h*)p;
      Aa[mt].h[1] = *(const v8h*)(p + 16);
    }
#pragma unroll
    for (int nt = 0; nt < 2; ++nt) {
      const _Float16* p = &vT[(nt * 16 + ln) * AT_STRIDE + 8 * g];
      Bv[nt].h[0] = *(const v8h*)p;
      Bv[nt].h[1] = *(const v8h*)(p + 16);
    }
    v8f yacc[2][2] = {};
#pragma unroll
    for (int mt = 0; mt < 2; ++mt)
#pragma unroll
      for (int nt = 0; nt < 2; ++nt)
        yacc[mt][nt] = wmma16(Aa[mt], Bv[nt], yacc[mt][nt]);

    __syncthreads();  // attnS reads complete; safe to reuse as spike staging
    // scale, LIF(0.5); spikes back into attnS[v][d] for coalesced writeout
#pragma unroll
    for (int mt = 0; mt < 2; ++mt)
#pragma unroll
      for (int nt = 0; nt < 2; ++nt)
#pragma unroll
        for (int r = 0; r < 8; ++r) {
          const int v = mt * 16 + 8 * g + r;
          const int d = nt * 16 + ln;
          float u = yacc[mt][nt][r] * 0.125f;
          float s = lif_step(vmem[mt][nt][r], u, 0.5f);
          attnS[v * AT_STRIDE + d] = (_Float16)s;
        }
    __syncthreads();

    // lane l writes row v=l: 64B contiguous (pad rows -> zeros)
    {
      _Float16* op = ys + base + (size_t)lane * C_;
#pragma unroll
      for (int dd = 0; dd < 4; ++dd) {
        v8h seg{};
        if (lane < V_) seg = *(const v8h*)&attnS[lane * AT_STRIDE + dd * 8];
        *(v8h*)(op + dd * 8) = (lane < V_) ? seg : v8h{};
      }
    }
    __syncthreads();
  }
}

// ---------------------------------------------------------------------------
// K3: per n: proj conv(+folded BN,bias) + LIF(1.0) + residual -> f32 out.
// Output staged in LDS [c][v] for fully coalesced residual-fused writeout.
// ---------------------------------------------------------------------------
__global__ __launch_bounds__(256) void proj_lif_kernel(
    const float* __restrict__ x, const _Float16* __restrict__ wp16,
    const float* __restrict__ bp2, const _Float16* __restrict__ ys,
    float* __restrict__ out) {
  __shared__ __align__(16) float Os[C_ * OS_STRIDE];

  const int n    = blockIdx.x;
  const int tid  = threadIdx.x;
  const int wave = tid >> 5;
  const int lane = tid & 31;
  const int g    = lane >> 4;
  const int ln   = lane & 15;
  const int m0[2] = { wave * 32, wave * 32 + 16 };

  float bias[2][8];
#pragma unroll
  for (int mt = 0; mt < 2; ++mt)
#pragma unroll
    for (int r = 0; r < 8; ++r) bias[mt][r] = bp2[m0[mt] + 8 * g + r];

  float vmem[2][2][8];
#pragma unroll
  for (int mt = 0; mt < 2; ++mt)
#pragma unroll
    for (int nt = 0; nt < 2; ++nt)
#pragma unroll
      for (int r = 0; r < 8; ++r) vmem[mt][nt][r] = 0.0f;

  for (int t = 0; t < T_; ++t) {
    const size_t ybase = (((size_t)t * N_ + n) * VP) * C_;
    if (t + 1 < T_) {
      const char* ny = (const char*)(ys + (((size_t)(t + 1) * N_ + n) * VP) * C_);
      __builtin_prefetch(ny + (size_t)tid * 64, 0, 1);
    }

    v8f acc[2][2] = {};
#pragma unroll
    for (int ks = 0; ks < C_ / 32; ++ks) {
      const int kb = ks * 32;
      Frag16 Af[2], Bf[2];
#pragma unroll
      for (int mt = 0; mt < 2; ++mt) {
        const _Float16* wr = wp16 + (size_t)(m0[mt] + ln) * C_ + kb + 8 * g;
        Af[mt].h[0] = *(const v8h*)(wr);
        Af[mt].h[1] = *(const v8h*)(wr + 16);
      }
#pragma unroll
      for (int nt = 0; nt < 2; ++nt) {
        const _Float16* yr = ys + ybase + (size_t)(nt * 16 + ln) * C_ + kb + 8 * g;
        Bf[nt].h[0] = *(const v8h*)(yr);
        Bf[nt].h[1] = *(const v8h*)(yr + 16);
      }
#pragma unroll
      for (int mt = 0; mt < 2; ++mt)
#pragma unroll
        for (int nt = 0; nt < 2; ++nt)
          acc[mt][nt] = wmma16(Af[mt], Bf[nt], acc[mt][nt]);
    }

    __syncthreads();  // protect Os reuse across t iterations
#pragma unroll
    for (int mt = 0; mt < 2; ++mt) {
#pragma unroll
      for (int nt = 0; nt < 2; ++nt) {
        const int v = nt * 16 + ln;
#pragma unroll
        for (int r = 0; r < 8; ++r) {
          const int o = m0[mt] + 8 * g + r;
          float u = acc[mt][nt][r] + bias[mt][r];
          float s = lif_step(vmem[mt][nt][r], u, 1.0f);
          if (v < V_) Os[o * OS_STRIDE + v] = s;
        }
      }
    }
    __syncthreads();

    const size_t obase = (((size_t)t * N_ + n) * C_) * V_;
    for (int idx = tid; idx < C_ * V_; idx += 256) {
      int c = idx / V_, v = idx - c * V_;
      out[obase + idx] = Os[c * OS_STRIDE + v] + x[obase + idx];  // residual
    }
  }
}

// ---------------------------------------------------------------------------
extern "C" void kernel_launch(void* const* d_in, const int* in_sizes, int n_in,
                              void* d_out, int out_size, void* d_ws,
                              size_t ws_size, hipStream_t stream) {
  const float* x    = (const float*)d_in[0];
  const float* wq   = (const float*)d_in[1];
  const float* wk   = (const float*)d_in[2];
  const float* wv   = (const float*)d_in[3];
  const float* wp   = (const float*)d_in[4];
  const float* bp   = (const float*)d_in[5];
  const float* qg   = (const float*)d_in[6];
  const float* qb   = (const float*)d_in[7];
  const float* qm   = (const float*)d_in[8];
  const float* qvar = (const float*)d_in[9];
  const float* kg   = (const float*)d_in[10];
  const float* kbta = (const float*)d_in[11];
  const float* km   = (const float*)d_in[12];
  const float* kvar = (const float*)d_in[13];
  const float* vg   = (const float*)d_in[14];
  const float* vb   = (const float*)d_in[15];
  const float* vm   = (const float*)d_in[16];
  const float* vvar = (const float*)d_in[17];
  const float* pg   = (const float*)d_in[18];
  const float* pb   = (const float*)d_in[19];
  const float* pm   = (const float*)d_in[20];
  const float* pvar = (const float*)d_in[21];

  char* ws = (char*)d_ws;
  const size_t WSZ = (size_t)C_ * C_ * sizeof(_Float16);  // 128 KB per matrix
  _Float16* wq16 = (_Float16*)(ws + 0 * WSZ);
  _Float16* wk16 = (_Float16*)(ws + 1 * WSZ);
  _Float16* wv16 = (_Float16*)(ws + 2 * WSZ);
  _Float16* wp16 = (_Float16*)(ws + 3 * WSZ);
  float* bqf = (float*)(ws + 4 * WSZ);
  float* bkf = bqf + C_;
  float* bvf = bkf + C_;
  float* bpf = bvf + C_;
  const size_t TILE = (size_t)T_ * N_ * VP * C_ * sizeof(_Float16);  // 32 MB
  _Float16* xT = (_Float16*)(ws + (1u << 20));
  _Float16* sq = (_Float16*)(ws + (1u << 20) + 1 * TILE);
  _Float16* sk = (_Float16*)(ws + (1u << 20) + 2 * TILE);
  _Float16* sv = (_Float16*)(ws + (1u << 20) + 3 * TILE);
  _Float16* ys = (_Float16*)(ws + (1u << 20) + 4 * TILE);

  const int foldBlocks = (C_ * C_ + 255) / 256;
  fold_bn_kernel<<<foldBlocks, 256, 0, stream>>>(wq, qg, qb, qm, qvar, nullptr,
                                                 wq16, bqf);
  fold_bn_kernel<<<foldBlocks, 256, 0, stream>>>(wk, kg, kbta, km, kvar,
                                                 nullptr, wk16, bkf);
  fold_bn_kernel<<<foldBlocks, 256, 0, stream>>>(wv, vg, vb, vm, vvar, nullptr,
                                                 wv16, bvf);
  fold_bn_kernel<<<foldBlocks, 256, 0, stream>>>(wp, pg, pb, pm, pvar, bp,
                                                 wp16, bpf);

  transpose_x_kernel<<<dim3(N_, T_), 256, 0, stream>>>(x, xT);

  qkv_conv_lif_kernel<<<dim3(N_, 3), 256, 0, stream>>>(
      xT, wq16, wk16, wv16, bqf, bkf, bvf, sq, sk, sv);

  attn_lif_kernel<<<dim3(N_, H_), 32, 0, stream>>>(sq, sk, sv, ys);

  proj_lif_kernel<<<N_, 256, 0, stream>>>(x, wp16, bpf, ys, (float*)d_out);
}
